// PeerNet_72438918414785
// MI455X (gfx1250) — matrix operand order repaired and checked by reading, not compiled, not run on hardware
//
#include <hip/hip_runtime.h>
#include <hip/hip_bf16.h>

typedef __attribute__((ext_vector_type(16))) __bf16 v16bf;
typedef __attribute__((ext_vector_type(8)))  float  v8f;
typedef __attribute__((ext_vector_type(4)))  float  v4f;

#define BATCH 4096

// ---------------------------------------------------------------------------
// out = [relu]( A[M,K] @ W[N,K]^T + bias )  via v_wmma_f32_16x16x32_bf16.
// One wave computes a 16x32 output strip (two 16x16 tiles sharing the A
// fragment -> two back-to-back WMMAs per K-step). f32 inputs converted to
// bf16 on the fly; f32 accumulation.
// Fragment layouts used here (wave32, contiguous k for A and B):
//   A (16x32 bf16): lane l -> row m=l%16, h=l/16; elems 0..7 = K[16h..16h+7],
//                   elems 8..15 = K[16h+8..16h+15]
//   B (32x16 bf16): lane l -> col n=l%16, h=l/16; elems 0..15 = K[16h..16h+15]
//   C/D (16x16 f32): VGPR r, lane l -> row r+8*(l/16), col l%16
// ---------------------------------------------------------------------------
template <bool RELU, bool TRANS>
__global__ void peernet_gemm_wmma(const float* __restrict__ A,
                                  const float* __restrict__ W,
                                  const float* __restrict__ bias,
                                  float* __restrict__ out,
                                  int M, int N, int K) {
    const int wave = blockIdx.x * (blockDim.x >> 5) + (threadIdx.x >> 5);
    const int lane = threadIdx.x & 31;
    const int pairsN = N >> 5;                       // 16x32 strips per row
    const int totalStrips = (M >> 4) * pairsN;
    if (wave >= totalStrips) return;                 // wave-uniform; EXEC all-1s

    const int tm = wave / pairsN;
    const int tp = wave - tm * pairsN;
    const int rowBase = tm << 4;
    const int colBase = tp << 5;

    const int m  = lane & 15;                        // A-row / B-col in tile
    const int hg = lane >> 4;                        // lane-group half

    const float* __restrict__ arow = A + (size_t)(rowBase + m) * K;
    const float* __restrict__ w0   = W + (size_t)(colBase + m) * K;
    const float* __restrict__ w1   = W + (size_t)(colBase + 16 + m) * K;

    v8f c0 = {}, c1 = {};
    for (int k0 = 0; k0 < K; k0 += 32) {
        // Unconditional speculative prefetch of the next K-block of the
        // streaming operand: translation faults are silently dropped per the
        // ISA, so no branch is needed in the hot loop.
        __builtin_prefetch(arow + k0 + 32, 0, 1);

        v16bf a, b0, b1;
        const float* a0 = arow + k0 + 16 * hg;       // K[16h .. 16h+7]
        const float* a1 = arow + k0 + 16 * hg + 8;   // K[16h+8 .. 16h+15]
#pragma unroll
        for (int e = 0; e < 8; ++e) {
            a[e]     = (__bf16)a0[e];
            a[8 + e] = (__bf16)a1[e];
        }
        const float* p0 = w0 + k0 + 16 * hg;         // K[16h .. 16h+15]
        const float* p1 = w1 + k0 + 16 * hg;
#pragma unroll
        for (int e = 0; e < 16; ++e) {
            b0[e] = (__bf16)p0[e];
            b1[e] = (__bf16)p1[e];
        }
        c0 = __builtin_amdgcn_wmma_f32_16x16x32_bf16(false, a, false, b0,
                                                     (short)0, c0, false, false);
        c1 = __builtin_amdgcn_wmma_f32_16x16x32_bf16(false, a, false, b1,
                                                     (short)0, c1, false, false);
    }

    const float bv0 = bias[colBase + m];
    const float bv1 = bias[colBase + 16 + m];
#pragma unroll
    for (int r = 0; r < 8; ++r) {
        c0[r] += bv0;
        c1[r] += bv1;
        if (RELU) { c0[r] = fmaxf(c0[r], 0.0f); c1[r] = fmaxf(c1[r], 0.0f); }
    }

    if (TRANS) {
        // Feature-major (h1T): lane's 8 values per tile are 8 consecutive
        // rows of one output row -> two 16-byte stores per tile.
        const int col = colBase + m;
        float* d0 = out + (size_t)col * M        + rowBase + 8 * hg;
        float* d1 = out + (size_t)(col + 16) * M + rowBase + 8 * hg;
        *(v4f*)d0       = v4f{c0[0], c0[1], c0[2], c0[3]};
        *(v4f*)(d0 + 4) = v4f{c0[4], c0[5], c0[6], c0[7]};
        *(v4f*)d1       = v4f{c1[0], c1[1], c1[2], c1[3]};
        *(v4f*)(d1 + 4) = v4f{c1[4], c1[5], c1[6], c1[7]};
    } else {
        // Batch-major: rows stride N, scalar stores.
        float* drow = out + (size_t)(rowBase + 8 * hg) * N + colBase + m;
#pragma unroll
        for (int r = 0; r < 8; ++r) {
            drow[(size_t)r * N]      = c0[r];
            drow[(size_t)r * N + 16] = c1[r];
        }
    }
}

// ---------------------------------------------------------------------------
// PeerNet pr_transform: per feature column, replace each value with the mean
// of its 6 nearest batch values (|diff|, self included).
// Nearest-k of a scalar form a contiguous window in sorted order: one
// workgroup per feature does a bitonic sort of (value, orig-index) in LDS,
// then a 5-step two-pointer window expansion, then scatters row-major.
// ---------------------------------------------------------------------------
__global__ void peernet_pr_knn(const float* __restrict__ h1T,
                               float* __restrict__ pr, int F) {
    __shared__ float          vals[BATCH];
    __shared__ unsigned short idxs[BATCH];

    const int f = blockIdx.x;
    const float* __restrict__ col = h1T + (size_t)f * BATCH;

    for (int i = threadIdx.x; i < BATCH; i += blockDim.x) {
        vals[i] = col[i];
        idxs[i] = (unsigned short)i;
    }
    __syncthreads();

    for (int k = 2; k <= BATCH; k <<= 1) {
        for (int j = k >> 1; j > 0; j >>= 1) {
            for (int i = threadIdx.x; i < BATCH; i += blockDim.x) {
                const int ixj = i ^ j;
                if (ixj > i) {
                    const bool up = ((i & k) == 0);
                    const float vi = vals[i], vj = vals[ixj];
                    if ((vi > vj) == up) {
                        vals[i] = vj; vals[ixj] = vi;
                        const unsigned short t = idxs[i];
                        idxs[i] = idxs[ixj]; idxs[ixj] = t;
                    }
                }
            }
            __syncthreads();
        }
    }

    for (int p = threadIdx.x; p < BATCH; p += blockDim.x) {
        const float vp = vals[p];
        int lo = p, hi = p;
        float sum = vp;
#pragma unroll
        for (int t = 0; t < 5; ++t) {
            bool takeLo;
            if (lo == 0)              takeLo = false;
            else if (hi == BATCH - 1) takeLo = true;
            else takeLo = (vp - vals[lo - 1]) <= (vals[hi + 1] - vp);
            if (takeLo) { --lo; sum += vals[lo]; }
            else        { ++hi; sum += vals[hi]; }
        }
        pr[(size_t)idxs[p] * F + f] = sum * (1.0f / 6.0f);
    }
}

// ---------------------------------------------------------------------------
// Final tiny layer: out[4096,2] = h2[4096,64] @ Wo[2,64]^T + bo  (1 MFLOP)
// ---------------------------------------------------------------------------
__global__ void peernet_out_linear(const float* __restrict__ h2,
                                   const float* __restrict__ Wo,
                                   const float* __restrict__ bo,
                                   float* __restrict__ out) {
    const int idx = blockIdx.x * blockDim.x + threadIdx.x;
    if (idx >= BATCH * 2) return;
    const int row = idx >> 1;
    const int o   = idx & 1;
    const float* __restrict__ hr = h2 + row * 64;
    const float* __restrict__ wr = Wo + o * 64;
    float s = bo[o];
#pragma unroll
    for (int k = 0; k < 64; ++k) s = fmaf(hr[k], wr[k], s);
    out[idx] = s;
}

extern "C" void kernel_launch(void* const* d_in, const int* in_sizes, int n_in,
                              void* d_out, int out_size, void* d_ws, size_t ws_size,
                              hipStream_t stream) {
    const float* x   = (const float*)d_in[0];   // [4096,1024]
    const float* W1  = (const float*)d_in[1];   // [128,1024]
    const float* b1  = (const float*)d_in[2];   // [128]
    const float* Wpr = (const float*)d_in[3];   // [128,128]
    const float* bpr = (const float*)d_in[4];   // [128]
    const float* W2  = (const float*)d_in[5];   // [64,128]
    const float* b2  = (const float*)d_in[6];   // [64]
    const float* Wo  = (const float*)d_in[7];   // [2,64]
    const float* bo  = (const float*)d_in[8];   // [2]
    float* out = (float*)d_out;                 // [4096,2]

    // Workspace (floats): h1T | pr_raw | pr2 | h2  = 7 MB total
    float* h1T = (float*)d_ws;                       // [128,4096] feature-major
    float* prr = h1T + (size_t)128 * BATCH;          // [4096,128]
    float* pr2 = prr + (size_t)BATCH * 128;          // [4096,128]
    float* h2  = pr2 + (size_t)BATCH * 128;          // [4096,64]

    const int THREADS = 128;                         // 4 waves / block
    const int WPB = THREADS / 32;

    // 1) h1T = relu(x @ W1^T + b1), stored transposed for contiguous columns
    {
        const int strips = (BATCH / 16) * (128 / 32);   // 1024 waves
        peernet_gemm_wmma<true, true><<<(strips + WPB - 1) / WPB, THREADS, 0, stream>>>(
            x, W1, b1, h1T, BATCH, 128, 1024);
    }
    // 2) pr_raw = 6-NN batch-mean per feature (sorted-window algorithm)
    peernet_pr_knn<<<128, 1024, 0, stream>>>(h1T, prr, 128);
    // 3) pr2 = relu(pr_raw @ Wpr^T + bpr)
    {
        const int strips = (BATCH / 16) * (128 / 32);
        peernet_gemm_wmma<true, false><<<(strips + WPB - 1) / WPB, THREADS, 0, stream>>>(
            prr, Wpr, bpr, pr2, BATCH, 128, 128);
    }
    // 4) h2 = relu(pr2 @ W2^T + b2)
    {
        const int strips = (BATCH / 16) * (64 / 32);    // 512 waves
        peernet_gemm_wmma<true, false><<<(strips + WPB - 1) / WPB, THREADS, 0, stream>>>(
            pr2, W2, b2, h2, BATCH, 64, 128);
    }
    // 5) out = h2 @ Wo^T + bo
    peernet_out_linear<<<(BATCH * 2 + 255) / 256, 256, 0, stream>>>(h2, Wo, bo, out);
}
